// SparseMoELayer_14972255994098
// MI455X (gfx1250) — matrix operand order repaired and checked by reading, not compiled, not run on hardware
//
#include <hip/hip_runtime.h>

// ---------------------------------------------------------------------------
// Sparse MoE forward for MI455X (gfx1250, wave32, WMMA + async LDS copy).
//   T=4096 tokens, D=1024, F=4096, E=8 experts, top-2 routing.
// Pipeline (all on `stream`):
//   1) k_zero            : out = 0, counts = 0
//   2) k_cvt_x           : x f32 -> xb bf16 (once; gather becomes a byte copy)
//   3) k_transpose_bf16  : W1 -> W1t[e][f][d] bf16, W2 -> W2t[e][n][f] bf16
//                          (K-contiguous for WMMA B operand; 134MB, L2-resident)
//   4) k_router          : logits -> softmax -> top2 -> per-expert (tok,w) lists
//   5) k_moe_ffn         : grouped fused FFN, M=32 token tiles, bf16 WMMA,
//                          X tile staged via global_load_async_to_lds_b128,
//                          every B register load feeds 2 WMMAs, atomic scatter
// Workspace: 256B counts | tok_idx | tok_w | W1t | W2t | xb  (~142.5 MB)
// ---------------------------------------------------------------------------

typedef __bf16 bf16;
typedef __attribute__((ext_vector_type(8)))  __bf16 v8bf;
typedef __attribute__((ext_vector_type(16))) __bf16 v16bf;
typedef __attribute__((ext_vector_type(8)))  float  v8f;

constexpr int T     = 2 * 2048;   // tokens
constexpr int DM    = 1024;       // d_model
constexpr int DF    = 4096;       // d_ff
constexpr int NE    = 8;          // experts
constexpr int ROWS  = 32;         // token tile (M) -> 2x B-operand reuse
constexpr int TPE   = T / ROWS;   // max tiles per expert (128)
constexpr int CHUNK = 256;        // d_ff chunk per iteration (8 waves * 32)

// LDS row strides padded +16 halves (32B): stride mod 256B == 32, so the two
// 16-lane halves of an A-operand ds_load_b128 hit disjoint bank groups.
constexpr int XA_S = DM + 16;     // 1040
constexpr int HA_S = CHUNK + 16;  // 272

// -------------------------------- init ------------------------------------
__global__ void k_zero(float* __restrict__ out, int n, int* __restrict__ counts) {
  int i = (blockIdx.x * blockDim.x + threadIdx.x) * 4;
#pragma unroll
  for (int j = 0; j < 4; ++j)
    if (i + j < n) out[i + j] = 0.f;
  if (blockIdx.x == 0 && threadIdx.x < NE) counts[threadIdx.x] = 0;
}

// --------------------------- x f32 -> bf16 ---------------------------------
__global__ void k_cvt_x(const float* __restrict__ x, bf16* __restrict__ xb, int n) {
  int i = (blockIdx.x * blockDim.x + threadIdx.x) * 8;
  if (i + 7 < n) {
    float4 f0 = *(const float4*)(x + i);
    float4 f1 = *(const float4*)(x + i + 4);
    v8bf p;
    p[0] = (bf16)f0.x; p[1] = (bf16)f0.y; p[2] = (bf16)f0.z; p[3] = (bf16)f0.w;
    p[4] = (bf16)f1.x; p[5] = (bf16)f1.y; p[6] = (bf16)f1.z; p[7] = (bf16)f1.w;
    *(v8bf*)(xb + i) = p;
  }
}

// --------------------- transpose + f32 -> bf16 convert ---------------------
// src: [E][R][C] f32  ->  dst: [E][C][R] bf16
__global__ void __launch_bounds__(256) k_transpose_bf16(
    const float* __restrict__ src, bf16* __restrict__ dst, int R, int C) {
  __shared__ float tile[32][33];
  const int e = blockIdx.z;
  const float* s = src + (size_t)e * R * C;
  bf16*        d = dst + (size_t)e * R * C;
  const int c0 = blockIdx.x * 32, r0 = blockIdx.y * 32;
  const int tx = threadIdx.x, ty = threadIdx.y;  // 32 x 8
#pragma unroll
  for (int k = 0; k < 32; k += 8)
    tile[ty + k][tx] = s[(size_t)(r0 + ty + k) * C + (c0 + tx)];
  __syncthreads();
#pragma unroll
  for (int k = 0; k < 32; k += 8)
    d[(size_t)(c0 + ty + k) * R + (r0 + tx)] = (bf16)tile[tx][ty + k];
}

// ------------------------------- router ------------------------------------
// One wave per token: logits = x[t] @ Wg, softmax, top-2, renormalize,
// append (token, weight) into per-expert compacted lists.
__global__ void __launch_bounds__(256) k_router(
    const float* __restrict__ x, const float* __restrict__ Wg,
    int* __restrict__ counts, int* __restrict__ tok_idx,
    float* __restrict__ tok_w) {
  const int wave = threadIdx.x >> 5, lane = threadIdx.x & 31;
  const int t = blockIdx.x * 8 + wave;
  const float* xr = x + (size_t)t * DM;

  float acc[NE];
#pragma unroll
  for (int e = 0; e < NE; ++e) acc[e] = 0.f;
  for (int i = lane; i < DM; i += 32) {
    const float xv = xr[i];
    const float* wr = Wg + (size_t)i * NE;   // Wg row: 8 contiguous floats
#pragma unroll
    for (int e = 0; e < NE; ++e) acc[e] += xv * wr[e];
  }
#pragma unroll
  for (int e = 0; e < NE; ++e)
    for (int off = 16; off; off >>= 1)
      acc[e] += __shfl_xor(acc[e], off, 32);

  if (lane == 0) {
    float m = acc[0];
#pragma unroll
    for (int e = 1; e < NE; ++e) m = fmaxf(m, acc[e]);
    float ex[NE];
#pragma unroll
    for (int e = 0; e < NE; ++e) ex[e] = __expf(acc[e] - m);
    int i0 = 0; float v0 = ex[0];
#pragma unroll
    for (int e = 1; e < NE; ++e) if (ex[e] > v0) { v0 = ex[e]; i0 = e; }
    int i1 = (i0 == 0) ? 1 : 0; float v1 = ex[i1];
#pragma unroll
    for (int e = 0; e < NE; ++e)
      if (e != i0 && ex[e] > v1) { v1 = ex[e]; i1 = e; }
    const float inv = 1.f / (v0 + v1);   // softmax denom cancels in renorm
    int p0 = atomicAdd(&counts[i0], 1);
    tok_idx[i0 * T + p0] = t; tok_w[i0 * T + p0] = v0 * inv;
    int p1 = atomicAdd(&counts[i1], 1);
    tok_idx[i1 * T + p1] = t; tok_w[i1 * T + p1] = v1 * inv;
  }
}

// ----------------------------- fused FFN -----------------------------------
// ISA A operand layout (16x32 bf16): lane l -> M = l&15,
//   elems 0..7  = K: 8*(l>>4) .. +7,  elems 8..15 = K: 16+8*(l>>4) .. +7
static __device__ __forceinline__ v16bf ld_a16(const bf16* p) {
  v8bf lo = *(const v8bf*)(p);
  v8bf hi = *(const v8bf*)(p + 16);
  return __builtin_shufflevector(lo, hi, 0, 1, 2, 3, 4, 5, 6, 7,
                                 8, 9, 10, 11, 12, 13, 14, 15);
}

// Async global->LDS copy, 16B per lane. Per ISA 10.x/§15.18: the 24-bit
// instruction offset is added to BOTH the global address and the LDS address,
// so one LDS-base VGPR + one 64-bit address pair covers a whole segment.
#define ASYNC_CP16(OFS)                                                     \
  asm volatile("global_load_async_to_lds_b128 %0, %1, off offset:" #OFS     \
               :: "v"(lds_base), "v"(gsrc) : "memory")

__global__ void __launch_bounds__(256) k_moe_ffn(
    const bf16* __restrict__ xb,    // [T][DM] bf16
    const bf16* __restrict__ W1t,   // [E][DF][DM] bf16 (K=d contiguous)
    const float* __restrict__ b1,   // [E][DF]
    const bf16* __restrict__ W2t,   // [E][DM][DF] bf16 (K=f contiguous)
    const float* __restrict__ b2,   // [E][DM]
    const int* __restrict__ counts,
    const int* __restrict__ tok_idx,
    const float* __restrict__ tok_w,
    float* __restrict__ out) {
  __shared__ bf16  XA[ROWS * XA_S];    // 65 KB: gathered x tile, bf16, padded
  __shared__ bf16  HA[ROWS * HA_S];    // 17 KB: relu(x@W1+b1) chunk, padded
  __shared__ int   tokLds[ROWS];
  __shared__ float gwLds[ROWS];

  const int e    = blockIdx.x / TPE;
  const int tile = blockIdx.x % TPE;
  const int Ne   = counts[e];
  if (tile * ROWS >= Ne) return;       // uniform early-exit for empty tiles

  const int tid = threadIdx.x;
  const int wave = tid >> 5, lane = tid & 31;
  const int lmod = lane & 15, lhi = lane >> 4;

  if (tid < ROWS) {
    const int s = tile * ROWS + tid;
    if (s < Ne) { tokLds[tid] = tok_idx[e * T + s]; gwLds[tid] = tok_w[e * T + s]; }
    else        { tokLds[tid] = 0;                  gwLds[tid] = 0.f; }  // weight 0 kills pad rows
  }
  __syncthreads();

  { // gather X tile -> LDS [32][1040] bf16 via async DMA (pure byte copy)
    const int r  = tid >> 3;                 // 0..31
    const int c0 = (tid & 7) * (DM / 8);     // 128-half (256B) segment
    const bf16* gsrc = xb + (size_t)tokLds[r] * DM + c0;
    const unsigned lds_base = (unsigned)(uintptr_t)(XA + r * XA_S + c0);
    ASYNC_CP16(0);   ASYNC_CP16(16);  ASYNC_CP16(32);  ASYNC_CP16(48);
    ASYNC_CP16(64);  ASYNC_CP16(80);  ASYNC_CP16(96);  ASYNC_CP16(112);
    ASYNC_CP16(128); ASYNC_CP16(144); ASYNC_CP16(160); ASYNC_CP16(176);
    ASYNC_CP16(192); ASYNC_CP16(208); ASYNC_CP16(224); ASYNC_CP16(240);
    asm volatile("s_wait_asynccnt 0x0" ::: "memory");
  }
  __syncthreads();

  const size_t w1base = (size_t)e * DF * DM;
  const size_t w2base = (size_t)e * DM * DF;

  v8f acc[8][2];                       // 32x128 output slab per wave (128 VGPRs)
#pragma unroll
  for (int t8 = 0; t8 < 8; ++t8)
#pragma unroll
    for (int h = 0; h < 2; ++h)
#pragma unroll
      for (int v = 0; v < 8; ++v) acc[t8][h][v] = 0.f;

  const int nbase = wave * 128;        // out-column range owned by this wave

  for (int fc = 0; fc < DF; fc += CHUNK) {
    // ---- phase 1: H[:, fc + wave*32 .. +32] = relu(X @ W1 + b1) ----
    v8f h00, h01, h10, h11;            // [m-half][f-tile]
#pragma unroll
    for (int v = 0; v < 8; ++v) { h00[v] = 0.f; h01[v] = 0.f; h10[v] = 0.f; h11[v] = 0.f; }
    const int f0 = fc + wave * 32;
    for (int k0 = 0; k0 < DM; k0 += 32) {
      v16bf a0 = ld_a16(XA + lmod * XA_S + k0 + 8 * lhi);          // rows 0-15
      v16bf a1 = ld_a16(XA + (16 + lmod) * XA_S + k0 + 8 * lhi);   // rows 16-31
      // B layout: lane l -> N = l&15, K 8*(l>>4)..+7 and 16+8*(l>>4)..+7
      const bf16* bp = W1t + w1base + (size_t)(f0 + lmod) * DM + k0 + 8 * lhi;
      v16bf b0v = ld_a16(bp);
      v16bf b1v = ld_a16(bp + (size_t)16 * DM);
      h00 = __builtin_amdgcn_wmma_f32_16x16x32_bf16(false, a0, false, b0v, (short)0, h00, false, false);
      h10 = __builtin_amdgcn_wmma_f32_16x16x32_bf16(false, a1, false, b0v, (short)0, h10, false, false);
      h01 = __builtin_amdgcn_wmma_f32_16x16x32_bf16(false, a0, false, b1v, (short)0, h01, false, false);
      h11 = __builtin_amdgcn_wmma_f32_16x16x32_bf16(false, a1, false, b1v, (short)0, h11, false, false);
    }
    { // bias + relu + store bf16 chunk to LDS
      const float bias0 = b1[(size_t)e * DF + f0 + lmod];
      const float bias1 = b1[(size_t)e * DF + f0 + 16 + lmod];
      const int colbase = wave * 32;
#pragma unroll
      for (int v = 0; v < 8; ++v) {
        const int m = v + 8 * lhi;     // C/D layout: VGPR v, lane-half -> M
        float q;
        q = h00[v] + bias0; q = q > 0.f ? q : 0.f; HA[m * HA_S + colbase + lmod]             = (bf16)q;
        q = h01[v] + bias1; q = q > 0.f ? q : 0.f; HA[m * HA_S + colbase + 16 + lmod]        = (bf16)q;
        q = h10[v] + bias0; q = q > 0.f ? q : 0.f; HA[(m + 16) * HA_S + colbase + lmod]      = (bf16)q;
        q = h11[v] + bias1; q = q > 0.f ? q : 0.f; HA[(m + 16) * HA_S + colbase + 16 + lmod] = (bf16)q;
      }
    }
    __syncthreads();

    // ---- phase 2: acc += H_chunk @ W2[fc..fc+CHUNK, nbase..nbase+128) ----
    for (int kk = 0; kk < CHUNK; kk += 32) {
      v16bf a0 = ld_a16(HA + lmod * HA_S + kk + 8 * lhi);          // A reused 8x
      v16bf a1 = ld_a16(HA + (16 + lmod) * HA_S + kk + 8 * lhi);
#pragma unroll
      for (int t8 = 0; t8 < 8; ++t8) {
        const int n = nbase + t8 * 16 + lmod;
        const bf16* bp = W2t + w2base + (size_t)n * DF + fc + kk + 8 * lhi;
        v16bf bv = ld_a16(bp);                                     // B reused 2x
        acc[t8][0] = __builtin_amdgcn_wmma_f32_16x16x32_bf16(false, a0, false, bv, (short)0, acc[t8][0], false, false);
        acc[t8][1] = __builtin_amdgcn_wmma_f32_16x16x32_bf16(false, a1, false, bv, (short)0, acc[t8][1], false, false);
      }
    }
    __syncthreads();                   // HA reused next iteration
  }

  // ---- weighted scatter-add: out[tok] += gate * (acc + b2) ----
#pragma unroll
  for (int t8 = 0; t8 < 8; ++t8) {
    const int n = nbase + t8 * 16 + lmod;
    const float bias = b2[(size_t)e * DM + n];
#pragma unroll
    for (int h = 0; h < 2; ++h)
#pragma unroll
      for (int v = 0; v < 8; ++v) {
        const int m = v + 8 * lhi + 16 * h;
        const float val = gwLds[m] * (acc[t8][h][v] + bias);
        __hip_atomic_fetch_add(out + (size_t)tokLds[m] * DM + n, val,
                               __ATOMIC_RELAXED, __HIP_MEMORY_SCOPE_AGENT);
      }
  }
}

// ------------------------------- launch ------------------------------------
extern "C" void kernel_launch(void* const* d_in, const int* in_sizes, int n_in,
                              void* d_out, int out_size, void* d_ws, size_t ws_size,
                              hipStream_t stream) {
  const float* x  = (const float*)d_in[0];
  const float* Wg = (const float*)d_in[1];
  const float* W1 = (const float*)d_in[2];
  const float* b1 = (const float*)d_in[3];
  const float* W2 = (const float*)d_in[4];
  const float* b2 = (const float*)d_in[5];
  float* out = (float*)d_out;

  // workspace layout (32B-aligned sections)
  char*  ws      = (char*)d_ws;
  int*   counts  = (int*)ws;                                   // 8 ints (256B pad)
  int*   tok_idx = (int*)(ws + 256);                           // [E][T]
  float* tok_w   = (float*)(ws + 256 + (size_t)NE * T * 4);    // [E][T]
  bf16*  W1t     = (bf16*)(ws + 256 + (size_t)NE * T * 8);     // [E][DF][DM]
  bf16*  W2t     = W1t + (size_t)NE * DF * DM;                 // [E][DM][DF]
  bf16*  xb      = W2t + (size_t)NE * DM * DF;                 // [T][DM]

  k_zero<<<(out_size + 1023) / 1024, 256, 0, stream>>>(out, out_size, counts);
  k_cvt_x<<<(T * DM) / (256 * 8), 256, 0, stream>>>(x, xb, T * DM);
  k_transpose_bf16<<<dim3(DF / 32, DM / 32, NE), dim3(32, 8), 0, stream>>>(
      W1, W1t, DM, DF);
  k_transpose_bf16<<<dim3(DM / 32, DF / 32, NE), dim3(32, 8), 0, stream>>>(
      W2, W2t, DF, DM);
  k_router<<<T / 8, 256, 0, stream>>>(x, Wg, counts, tok_idx, tok_w);
  k_moe_ffn<<<NE * TPE, 256, 0, stream>>>(
      xb, W1t, b1, W2t, b2, counts, tok_idx, tok_w, out);
}
